// EncoderNetwork_12979391169438
// MI455X (gfx1250) — matrix operand order repaired and checked by reading, not compiled, not run on hardware
//
#include <hip/hip_runtime.h>
#include <hip/hip_bf16.h>

typedef __bf16 bf16_t;
typedef __attribute__((ext_vector_type(16))) __bf16 v16bf;
typedef __attribute__((ext_vector_type(8)))  __bf16 v8bf;
typedef __attribute__((ext_vector_type(8)))  float  v8f;

#define BM 128
#define BN 128
#define BK 32

union Frag  { v16bf v; v8bf h[2]; };
union CvU   { bf16_t e[16]; uint4 q[2]; };

static inline int cdiv(long a, long b){ return (int)((a + b - 1) / b); }

// ---------------------------------------------------------------------------
// GEMM: C[M,Nout] = act(A[M,K] @ W + bias)
//   - K % 32 == 0 (inputs pre-padded), Nout % 128 == 0 (128 or 256 here)
//   - Wt is [Nout, K] bf16, pre-transposed (k-major per row)
//   - A rows clamped (not zeroed) at the M edge: LDS row r only feeds output
//     row r, and rows >= M are never stored.
// Block: 256 thr = 8 waves; tile 128x128; wave = 32x64 tile (2x4 WMMA accum).
// Double-buffered LDS, ONE barrier per K-chunk:
//   issue global loads (k+1) -> 8 WMMAs on buf(k) -> cvt+store buf(k^1) -> bar
// ---------------------------------------------------------------------------
__global__ __launch_bounds__(256)
void k_gemm(const float* __restrict__ A, const bf16_t* __restrict__ Wt,
            const float* __restrict__ bias, float* __restrict__ C,
            int M, int K, int Nout, int relu)
{
  __shared__ __align__(16) bf16_t As[2][BM * BK];   // [buf][row][k]  2x8 KB
  __shared__ __align__(16) bf16_t Bs[2][BN * BK];   // [buf][col][k]  2x8 KB

  const int tid  = threadIdx.x;
  const int lane = tid & 31;
  const int wave = tid >> 5;
  const int wrow = wave & 3;      // 4 row-groups of 32 rows
  const int wcol = wave >> 2;     // 2 col-groups of 64 cols
  const int rowbase = blockIdx.x * BM;
  const int colbase = blockIdx.y * BN;

  // staging assignments (uniform, branch-free)
  const int arow = tid >> 1, ahalf = tid & 1;
  int gr = rowbase + arow; if (gr > M - 1) gr = M - 1;     // clamp, never zero
  const float*  Arow = A  + (size_t)gr * K + ahalf * 16;
  const int bcol = tid >> 1, bpart = tid & 1;
  const bf16_t* Brow = Wt + (size_t)(colbase + bcol) * K + bpart * 16;

  const int aoff = arow * BK + ahalf * 16;   // LDS store offsets (elements)
  const int boff = bcol * BK + bpart * 16;

  // per-lane fragment decomposition (ISA 7.12.2, wave32)
  const int fm  = lane & 15;
  const int akb = (lane < 16) ? 0 : 8;    // A frag k-base
  const int bkb = (lane < 16) ? 0 : 8;    // B frag k-base

  v8f acc[2][4] = {};

  float4 a4[4];
  uint4  bq0, bq1;

  // ---- prologue: load + stage chunk 0 into buf 0 ----
  {
    const float4* ap = (const float4*)Arow;
    a4[0] = ap[0]; a4[1] = ap[1]; a4[2] = ap[2]; a4[3] = ap[3];
    const uint4* bp = (const uint4*)Brow;
    bq0 = bp[0]; bq1 = bp[1];

    CvU cv;
    #pragma unroll
    for (int q = 0; q < 4; ++q) {
      cv.e[q * 4 + 0] = (bf16_t)a4[q].x;
      cv.e[q * 4 + 1] = (bf16_t)a4[q].y;
      cv.e[q * 4 + 2] = (bf16_t)a4[q].z;
      cv.e[q * 4 + 3] = (bf16_t)a4[q].w;
    }
    uint4* asd = (uint4*)&As[0][aoff];
    asd[0] = cv.q[0]; asd[1] = cv.q[1];
    uint4* bsd = (uint4*)&Bs[0][boff];
    bsd[0] = bq0; bsd[1] = bq1;
  }
  __syncthreads();

  int buf = 0;
  for (int kp = 0; kp < K; kp += BK) {
    const bool next = (kp + BK) < K;

    // ---- issue global loads for chunk k+1 (overlap with WMMAs below) ----
    if (next) {
      const float4* ap = (const float4*)(Arow + kp + BK);
      a4[0] = ap[0]; a4[1] = ap[1]; a4[2] = ap[2]; a4[3] = ap[3];
      const uint4* bp = (const uint4*)(Brow + kp + BK);
      bq0 = bp[0]; bq1 = bp[1];
    }

    // ---- fragments + 8 WMMAs on current buffer ----
    const bf16_t* Ab = As[buf];
    const bf16_t* Bb = Bs[buf];
    Frag fa[2], fb[4];
    #pragma unroll
    for (int i = 0; i < 2; ++i) {
      const bf16_t* ab = &Ab[(wrow * 32 + i * 16 + fm) * BK];
      fa[i].h[0] = *(const v8bf*)(ab + akb);
      fa[i].h[1] = *(const v8bf*)(ab + akb + 16);
    }
    #pragma unroll
    for (int j = 0; j < 4; ++j) {
      const bf16_t* bb = &Bb[(wcol * 64 + j * 16 + fm) * BK + bkb];
      fb[j].h[0] = *(const v8bf*)(bb);
      fb[j].h[1] = *(const v8bf*)(bb + 16);
    }
    #pragma unroll
    for (int i = 0; i < 2; ++i)
      #pragma unroll
      for (int j = 0; j < 4; ++j)
        acc[i][j] = __builtin_amdgcn_wmma_f32_16x16x32_bf16(
            false, fa[i].v, false, fb[j].v, (short)0, acc[i][j], false, false);

    // ---- convert + store chunk k+1 into the other buffer ----
    // (buf^1 was last read in the previous iteration; the single barrier
    //  below already separated those reads from these writes)
    if (next) {
      CvU cv;
      #pragma unroll
      for (int q = 0; q < 4; ++q) {
        cv.e[q * 4 + 0] = (bf16_t)a4[q].x;
        cv.e[q * 4 + 1] = (bf16_t)a4[q].y;
        cv.e[q * 4 + 2] = (bf16_t)a4[q].z;
        cv.e[q * 4 + 3] = (bf16_t)a4[q].w;
      }
      uint4* asd = (uint4*)&As[buf ^ 1][aoff];
      asd[0] = cv.q[0]; asd[1] = cv.q[1];
      uint4* bsd = (uint4*)&Bs[buf ^ 1][boff];
      bsd[0] = bq0; bsd[1] = bq1;
    }
    __syncthreads();
    buf ^= 1;
  }

  // ---- epilogue: bias + optional relu (stores coalesce per half-wave) ----
  const int rlo = (lane < 16) ? 0 : 8;
  const bool fullTile = (rowbase + BM) <= M;
  #pragma unroll
  for (int j = 0; j < 4; ++j) {
    const int gcol = colbase + wcol * 64 + j * 16 + fm;
    const float bv = bias[gcol];
    #pragma unroll
    for (int i = 0; i < 2; ++i) {
      const int rb = rowbase + wrow * 32 + i * 16 + rlo;
      if (fullTile) {
        #pragma unroll
        for (int r = 0; r < 8; ++r) {
          float v = acc[i][j][r] + bv;
          if (relu) v = fmaxf(v, 0.0f);
          C[(size_t)(rb + r) * Nout + gcol] = v;
        }
      } else {
        #pragma unroll
        for (int r = 0; r < 8; ++r) {
          if (rb + r < M) {
            float v = acc[i][j][r] + bv;
            if (relu) v = fmaxf(v, 0.0f);
            C[(size_t)(rb + r) * Nout + gcol] = v;
          }
        }
      }
    }
  }
}

// ---------------------------------------------------------------------------
// Helper kernels
// ---------------------------------------------------------------------------
__global__ void k_wconv(const float* __restrict__ W, bf16_t* __restrict__ Wt,
                        int din, int dout, int Kp)
{
  int idx = blockIdx.x * 256 + threadIdx.x;
  if (idx >= dout * Kp) return;
  int n = idx / Kp, k = idx - n * Kp;
  float v = (k < din) ? W[(size_t)k * dout + n] : 0.0f;   // transpose + pad
  Wt[idx] = (bf16_t)v;
}

// x [N,8] -> xpad [N,32] zero-padded
__global__ void k_padx(const float* __restrict__ x, float* __restrict__ xp, long n){
  long i = (long)blockIdx.x * 256 + threadIdx.x;
  if (i < n) {
    long node = i >> 5; int c = (int)(i & 31);
    xp[i] = (c < 8) ? x[node * 8 + c] : 0.0f;
  }
}

__global__ void k_zero_f(float* p, long n){ long i=(long)blockIdx.x*256+threadIdx.x; if(i<n) p[i]=0.0f; }
__global__ void k_zero_i(int*   p, int  n){ int  i=blockIdx.x*256+threadIdx.x;       if(i<n) p[i]=0;    }
__global__ void k_copy_f(const float* a, float* b, long n){ long i=(long)blockIdx.x*256+threadIdx.x; if(i<n) b[i]=a[i]; }

__global__ void k_mark(const int* __restrict__ idx, int* __restrict__ flag, int E){
  int e = blockIdx.x * 256 + threadIdx.x;
  if (e < E) flag[idx[e]] = 1;
}

// h = has_in ? 0 : update(h_init)
__global__ void k_h0(const float* __restrict__ upd, const int* __restrict__ has_in,
                     float* __restrict__ h, long n){
  long i = (long)blockIdx.x * 256 + threadIdx.x;
  if (i < n) h[i] = has_in[i >> 7] ? 0.0f : upd[i];
}

__global__ void k_dmask(const unsigned char* __restrict__ m, const int* __restrict__ dst,
                        int* __restrict__ dm, int E){
  int e = blockIdx.x * 256 + threadIdx.x;
  if (e < E && m[e]) dm[dst[e]] = 1;
}

// agg[dst] += msg[src] over masked edges; one block per edge, 128 lanes = D
__global__ __launch_bounds__(128)
void k_scatter(const unsigned char* __restrict__ m, const int* __restrict__ src,
               const int* __restrict__ dst, const float* __restrict__ msg,
               float* __restrict__ agg, int E){
  int e = blockIdx.x;
  if (e >= E || !m[e]) return;
  int d = threadIdx.x;
  atomicAdd(&agg[(size_t)dst[e] * 128 + d], msg[(size_t)src[e] * 128 + d]);
}

// h = dst_mask ? h_init + update(agg) : h
__global__ void k_comb(const float* __restrict__ hi, const float* __restrict__ ub,
                       const int* __restrict__ dm, float* __restrict__ h, long n){
  long i = (long)blockIdx.x * 256 + threadIdx.x;
  if (i < n && dm[i >> 7]) h[i] = hi[i] + ub[i];
}

// cat [N,160]: [ x(8) | h(128) | zeros(24) ]
__global__ void k_cat(const float* __restrict__ x, const float* __restrict__ h,
                      float* __restrict__ cat, long n){
  long i = (long)blockIdx.x * 256 + threadIdx.x;
  if (i < n) {
    long node = i / 160; int c = (int)(i - node * 160);
    float v = 0.0f;
    if (c < 8)        v = x[node * 8 + c];
    else if (c < 136) v = h[node * 128 + (c - 8)];
    cat[i] = v;
  }
}

// CSR segment sum over rows of a [*,128] matrix; one block per segment
__global__ __launch_bounds__(128)
void k_segsum(const float* __restrict__ v, const int* __restrict__ ptr,
              float* __restrict__ out){
  int g = blockIdx.x, d = threadIdx.x;
  int r0 = ptr[g], r1 = ptr[g + 1];
  float s = 0.0f;
  for (int r = r0; r < r1; ++r) s += v[(size_t)r * 128 + d];
  out[(size_t)g * 128 + d] = s;
}

// ---------------------------------------------------------------------------
extern "C" void kernel_launch(void* const* d_in, const int* in_sizes, int n_in,
                              void* d_out, int out_size, void* d_ws, size_t ws_size,
                              hipStream_t stream)
{
  (void)n_in; (void)out_size; (void)ws_size;
  const int Fdim = 8, Dm = 128, Hm = 256;
  const int N    = in_sizes[0] / Fdim;
  const int E    = in_sizes[1] / 2;
  const int L    = in_sizes[2] / E;
  const int NDAG = in_sizes[3] - 1;
  const int NOBS = in_sizes[4] - 1;

  const float* x = (const float*)d_in[0];
  const int*   ei  = (const int*)d_in[1];
  const int*   dst = ei;          // edge_index[0] (message destination)
  const int*   src = ei + E;      // edge_index[1] (message source)
  const unsigned char* emask = (const unsigned char*)d_in[2];  // bool [L,E]
  const int*   ptr  = (const int*)d_in[3];
  const int*   optr = (const int*)d_in[4];

  struct Mlp { const float* W[3]; const float* b[3]; bf16_t* Wt[3];
               int din[3], dout[3], Kp[3]; };

  uintptr_t wp = (uintptr_t)d_ws;
  auto alloc = [&](size_t bytes) -> void* {
    void* r = (void*)wp;
    wp += (bytes + 255) & ~(size_t)255;
    return r;
  };

  auto mkMlp = [&](int base, int k0) -> Mlp {
    Mlp m;
    int dims[4] = {k0, Hm, Hm, Dm};
    for (int l = 0; l < 3; ++l) {
      m.W[l] = (const float*)d_in[base + 2 * l];
      m.b[l] = (const float*)d_in[base + 2 * l + 1];
      m.din[l] = dims[l]; m.dout[l] = dims[l + 1];
      m.Kp[l]  = ((dims[l] + 31) / 32) * 32;
      m.Wt[l]  = (bf16_t*)alloc((size_t)m.dout[l] * m.Kp[l] * sizeof(bf16_t));
    }
    return m;
  };

  // params pytree flattened in sorted-key order: dag, glob, msg, prep, update
  Mlp mDag  = mkMlp(5,  Fdim + Dm);
  Mlp mGlob = mkMlp(11, Dm);
  Mlp mMsg  = mkMlp(17, Dm);
  Mlp mPrep = mkMlp(23, Fdim);
  Mlp mUpd  = mkMlp(29, Dm);

  Mlp* ms[5] = {&mDag, &mGlob, &mMsg, &mPrep, &mUpd};
  for (int i = 0; i < 5; ++i)
    for (int l = 0; l < 3; ++l) {
      int n = ms[i]->dout[l] * ms[i]->Kp[l];
      k_wconv<<<cdiv(n, 256), 256, 0, stream>>>(ms[i]->W[l], ms[i]->Wt[l],
                                                ms[i]->din[l], ms[i]->dout[l], ms[i]->Kp[l]);
    }

  float* t0     = (float*)alloc((size_t)N * Hm * sizeof(float));
  float* t1     = (float*)alloc((size_t)N * Hm * sizeof(float));
  float* h_init = (float*)alloc((size_t)N * Dm * sizeof(float));
  float* h      = (float*)alloc((size_t)N * Dm * sizeof(float));
  float* ubuf   = (float*)alloc((size_t)N * Dm * sizeof(float));
  float* msgbuf = (float*)alloc((size_t)N * Dm * sizeof(float));
  float* agg    = (float*)alloc((size_t)N * Dm * sizeof(float));
  float* xpad   = (float*)alloc((size_t)N * 32 * sizeof(float));
  float* catb   = (float*)alloc((size_t)N * 160 * sizeof(float));
  int*   has_in = (int*)alloc((size_t)N * sizeof(int));
  int*   dmask  = (int*)alloc((size_t)N * sizeof(int));

  float* out_node = (float*)d_out;
  float* out_dag  = out_node + (size_t)N * Dm;
  float* out_glob = out_dag  + (size_t)NDAG * Dm;

  auto gemm = [&](const float* A, const Mlp& m, int l, float* C, int M, int relu) {
    dim3 grid((unsigned)cdiv(M, BM), (unsigned)(m.dout[l] / BN));
    k_gemm<<<grid, 256, 0, stream>>>(A, m.Wt[l], m.b[l], C,
                                     M, m.Kp[l], m.dout[l], relu);
  };
  auto mlp3 = [&](const float* A, const Mlp& m, int M, float* out) {
    gemm(A,  m, 0, t0,  M, 1);
    gemm(t0, m, 1, t1,  M, 1);
    gemm(t1, m, 2, out, M, 0);
  };

  // ---- NodeEncoder ----
  k_padx<<<cdiv((long)N * 32, 256), 256, 0, stream>>>(x, xpad, (long)N * 32);
  mlp3(xpad,   mPrep, N, h_init);
  mlp3(h_init, mUpd,  N, ubuf);

  k_zero_i<<<cdiv(N, 256), 256, 0, stream>>>(has_in, N);
  k_mark<<<cdiv(E, 256), 256, 0, stream>>>(dst, has_in, E);
  k_h0<<<cdiv((long)N * Dm, 256), 256, 0, stream>>>(ubuf, has_in, h, (long)N * Dm);

  for (int lvl = L - 1; lvl >= 0; --lvl) {
    const unsigned char* m = emask + (size_t)lvl * E;
    k_zero_i<<<cdiv(N, 256), 256, 0, stream>>>(dmask, N);
    k_dmask<<<cdiv(E, 256), 256, 0, stream>>>(m, dst, dmask, E);

    mlp3(h, mMsg, N, msgbuf);  // src-mask drop is a no-op for agg

    k_zero_f<<<cdiv((long)N * Dm, 256), 256, 0, stream>>>(agg, (long)N * Dm);
    k_scatter<<<E, 128, 0, stream>>>(m, src, dst, msgbuf, agg, E);

    mlp3(agg, mUpd, N, ubuf);
    k_comb<<<cdiv((long)N * Dm, 256), 256, 0, stream>>>(h_init, ubuf, dmask, h, (long)N * Dm);
  }

  k_copy_f<<<cdiv((long)N * Dm, 256), 256, 0, stream>>>(h, out_node, (long)N * Dm);

  // ---- DagEncoder ----
  k_cat<<<cdiv((long)N * 160, 256), 256, 0, stream>>>(x, h, catb, (long)N * 160);
  mlp3(catb, mDag, N, msgbuf);                       // reuse msgbuf
  k_segsum<<<NDAG, 128, 0, stream>>>(msgbuf, ptr, out_dag);

  // ---- GlobalEncoder ----
  mlp3(out_dag, mGlob, NDAG, agg);                   // reuse agg head
  k_segsum<<<NOBS, 128, 0, stream>>>(agg, optr, out_glob);
}
